// cnn_encoder_66975720013831
// MI455X (gfx1250) — matrix operand order repaired and checked by reading, not compiled, not run on hardware
//
#include <hip/hip_runtime.h>
#include <hip/hip_bf16.h>
#include <math.h>

// ---------------------------------------------------------------------------
// Types for CDNA5 WMMA (wave32): v_wmma_f32_16x16x32_bf16
// ---------------------------------------------------------------------------
typedef __bf16 bf16_t;
typedef __attribute__((ext_vector_type(16))) __bf16        v16bf;
typedef __attribute__((ext_vector_type(8)))  float         v8f;
typedef __attribute__((ext_vector_type(4)))  unsigned int  v4u;

union Frag16 { v16bf v; v4u q[2]; };

// ---------------------------------------------------------------------------
// Block 1: direct conv (Cin=3 too thin for WMMA). x: NCDHW fp32 [8,3,64,64,64]
// out: NDHWC fp32 [8,32,32,32,32], stride 2, pad 1.
// ---------------------------------------------------------------------------
__launch_bounds__(256)
__global__ void conv1_direct_kernel(const float* __restrict__ x,
                                    const float* __restrict__ w1,
                                    float* __restrict__ out, long total)
{
    for (long e = (long)blockIdx.x * blockDim.x + threadIdx.x; e < total;
         e += (long)gridDim.x * blockDim.x) {
        int  c = (int)(e & 31);
        long s = e >> 5;                 // b*32768 + site
        int  b = (int)(s >> 15);
        int  r = (int)(s & 32767);
        int  z = r >> 10, y = (r >> 5) & 31, xx = r & 31;
        float acc = 0.f;
        for (int kd = 0; kd < 3; ++kd) {
            int iz = 2 * z + kd - 1; if (iz < 0 || iz >= 64) continue;
            for (int kh = 0; kh < 3; ++kh) {
                int iy = 2 * y + kh - 1; if (iy < 0 || iy >= 64) continue;
                for (int kw = 0; kw < 3; ++kw) {
                    int ix = 2 * xx + kw - 1; if (ix < 0 || ix >= 64) continue;
                    #pragma unroll
                    for (int i = 0; i < 3; ++i) {
                        float xv = x[(((long)(b * 3 + i) * 64 + iz) * 64 + iy) * 64 + ix];
                        float wv = w1[(((c * 3 + i) * 3 + kd) * 3 + kh) * 3 + kw];
                        acc = fmaf(xv, wv, acc);
                    }
                }
            }
        }
        out[e] = acc;
    }
}

// ---------------------------------------------------------------------------
// Weight repack: w fp32 [Cout][Cin][3][3][3] -> bf16 fragments in exact
// per-lane B order this file uses for v_wmma_f32_16x16x32_bf16 (contiguous K, same as A).
// Fragment (koff, cb, nb): 32 lanes x 8 dwords (2 bf16 each).
// Lane p (0..15): col n=p, K = 0..15 ; lane p+16: col n=p, K = 16..31.
// dword dw holds K = phase + 2*dw, 2*dw+1.
// ---------------------------------------------------------------------------
__launch_bounds__(256)
__global__ void repack_w_kernel(const float* __restrict__ w,
                                bf16_t* __restrict__ outp,
                                int Cin, int Cout, long total /*dwords*/)
{
    int coutB = Cout >> 4, cinB = Cin >> 5;
    for (long e = (long)blockIdx.x * blockDim.x + threadIdx.x; e < total;
         e += (long)gridDim.x * blockDim.x) {
        int  dw   = (int)(e & 7);
        int  lane = (int)((e >> 3) & 31);
        long frag = e >> 8;
        int  nb = (int)(frag % coutB);
        long t  = frag / coutB;
        int  cb = (int)(t % cinB);
        int  koff = (int)(t / cinB);
        int  p = lane & 15, hi = lane >> 4;
        int  kk = hi * 16 + dw * 2;
        int  i0 = cb * 32 + kk;
        int  o  = nb * 16 + p;
        float f0 = w[((long)o * Cin + i0)     * 27 + koff];
        float f1 = w[((long)o * Cin + i0 + 1) * 27 + koff];
        outp[2 * e]     = (bf16_t)f0;
        outp[2 * e + 1] = (bf16_t)f1;
    }
}

// ---------------------------------------------------------------------------
// Implicit-GEMM 3x3x3 conv via WMMA, shape-specialized (compile-time dims ->
// shift/mask address math, unrollable K loop). One wave = 16 sites x 32
// out-channels (dual accumulator). Unroll bounded to keep codegen tractable.
// in: bf16 NDHWC; wp: repacked bf16 fragments; out: fp32 NDHWC.
// ---------------------------------------------------------------------------
template <int DI, int HI, int WI, int CIN, int COUT, int STRIDE>
__launch_bounds__(256)
__global__ void conv3d_wmma_kernel(const bf16_t* __restrict__ in,
                                   const bf16_t* __restrict__ wp,
                                   float* __restrict__ out, int Bn)
{
    constexpr int CINB  = CIN / 32;     // 32-channel K blocks
    constexpr int COUTB = COUT / 16;    // 16-col N blocks
    constexpr int NT    = COUT / 32;    // dual-N tiles per site-tile
    constexpr int DO    = (DI - 1) / STRIDE + 1;   // pad=1, k=3
    constexpr int HO    = (HI - 1) / STRIDE + 1;
    constexpr int WO    = (WI - 1) / STRIDE + 1;
    constexpr int NS    = DO * HO * WO;

    const int lane = threadIdx.x & 31;
    const int wave = threadIdx.x >> 5;
    const int mTiles = (Bn * NS) >> 4;
    const long tileId = (long)blockIdx.x * 8 + wave;
    if (tileId >= (long)mTiles * NT) return;     // wave-uniform exit

    const int nb2 = (int)(tileId % NT);          // which 32-col pair
    const int tm  = (int)(tileId / NT);
    const int p   = lane & 15;
    const int hi  = lane >> 4;

    // decompose this lane's A-row site (NS, HO*WO, WO are compile-time)
    const int s = tm * 16 + p;
    const int b = s / NS;
    int r = s - b * NS;
    const int z  = r / (HO * WO); r -= z * HO * WO;
    const int y  = r / WO;
    const int xx = r - y * WO;

    v8f acc0 = {0.f, 0.f, 0.f, 0.f, 0.f, 0.f, 0.f, 0.f};
    v8f acc1 = {0.f, 0.f, 0.f, 0.f, 0.f, 0.f, 0.f, 0.f};
    const v4u zero = {0u, 0u, 0u, 0u};

    for (int kd = 0; kd < 3; ++kd) {
        const int iz = z * STRIDE + kd - 1;
        for (int kh = 0; kh < 3; ++kh) {
            const int iy = y * STRIDE + kh - 1;
            for (int kw = 0; kw < 3; ++kw) {
                const int ix = xx * STRIDE + kw - 1;
                const bool valid = (iz >= 0) & (iz < DI) & (iy >= 0) & (iy < HI) &
                                   (ix >= 0) & (ix < WI);
                const long so = ((((long)b * DI + iz) * HI + iy) * WI + ix) * CIN;
                const int  koff = (kd * 3 + kh) * 3 + kw;
                // two adjacent 16-col fragments -> contiguous 2KB
                const bf16_t* wf = wp + ((long)(koff * CINB) * COUTB + nb2 * 2) * 512
                                      + lane * 16;
                #pragma unroll 2
                for (int cb = 0; cb < CINB; ++cb) {
                    Frag16 a, b0, b1;
                    if (valid) {
                        const bf16_t* ap = in + so + cb * 32 + hi * 16;
                        a.q[0] = *(const v4u*)(ap);
                        a.q[1] = *(const v4u*)(ap + 8);
                    } else {
                        a.q[0] = zero; a.q[1] = zero;
                    }
                    const v4u* bp4 = (const v4u*)(wf + (long)cb * COUTB * 512);
                    b0.q[0] = bp4[0];
                    b0.q[1] = bp4[1];
                    b1.q[0] = bp4[32];   // +256 bf16 elems = +32 v4u
                    b1.q[1] = bp4[33];
                    acc0 = __builtin_amdgcn_wmma_f32_16x16x32_bf16(
                        false, a.v, false, b0.v, (short)0, acc0, false, false);
                    acc1 = __builtin_amdgcn_wmma_f32_16x16x32_bf16(
                        false, a.v, false, b1.v, (short)0, acc1, false, false);
                }
            }
        }
    }

    // D layout: lane p(/p+16): col n0+p, VGPR j -> row hi*8 + j
    const int n0 = nb2 * 32;
    float* op = out + ((long)tm * 16 + hi * 8) * COUT + n0 + p;
    #pragma unroll
    for (int j = 0; j < 8; ++j) {
        op[(long)j * COUT]      = acc0[j];
        op[(long)j * COUT + 16] = acc1[j];
    }
}

// ---------------------------------------------------------------------------
// Instance-norm stats: one block per (b,c); fp32 NDHWC input.
// stats[2*bc] = mean, stats[2*bc+1] = rsqrt(var + eps)
// ---------------------------------------------------------------------------
__launch_bounds__(256)
__global__ void in_stats_kernel(const float* __restrict__ f,
                                float* __restrict__ stats, int Ns, int C)
{
    const int bc = blockIdx.x;
    const int b = bc / C, c = bc - b * C;
    const float* base = f + (long)b * Ns * C + c;
    float s = 0.f, sq = 0.f;
    for (int i = threadIdx.x; i < Ns; i += 256) {
        float v = base[(long)i * C];
        s += v; sq += v * v;
    }
    __shared__ float sh[512];
    sh[threadIdx.x] = s; sh[256 + threadIdx.x] = sq;
    __syncthreads();
    for (int o = 128; o; o >>= 1) {
        if (threadIdx.x < (unsigned)o) {
            sh[threadIdx.x]       += sh[threadIdx.x + o];
            sh[256 + threadIdx.x] += sh[256 + threadIdx.x + o];
        }
        __syncthreads();
    }
    if (threadIdx.x == 0) {
        float mean = sh[0] / (float)Ns;
        float var  = sh[256] / (float)Ns - mean * mean;
        if (var < 0.f) var = 0.f;
        stats[2 * bc]     = mean;
        stats[2 * bc + 1] = rsqrtf(var + 1e-5f);
    }
}

// ---------------------------------------------------------------------------
// Normalize + ELU + convert to bf16 (ELU is monotone, so ELU-then-maxpool
// equals the reference's maxpool-then-ELU).
// ---------------------------------------------------------------------------
__launch_bounds__(256)
__global__ void norm_elu_bf16_kernel(const float* __restrict__ f,
                                     const float* __restrict__ stats,
                                     bf16_t* __restrict__ out,
                                     long total, int Ns, int C)
{
    for (long e = (long)blockIdx.x * blockDim.x + threadIdx.x; e < total;
         e += (long)gridDim.x * blockDim.x) {
        int  c = (int)(e % C);
        long s = e / C;
        int  b = (int)(s / Ns);
        int  bc = b * C + c;
        float v = (f[e] - stats[2 * bc]) * stats[2 * bc + 1];
        v = v > 0.f ? v : expm1f(v);
        out[e] = (bf16_t)v;
    }
}

// ---------------------------------------------------------------------------
// Maxpool k=3 s=2 pad=1 (-inf padding == skip OOB), bf16 NDHWC.
// ---------------------------------------------------------------------------
__launch_bounds__(256)
__global__ void maxpool3_bf16_kernel(const bf16_t* __restrict__ in,
                                     bf16_t* __restrict__ out,
                                     int Bn, int Di, int Hi, int Wi, int C,
                                     long total)
{
    const int Do_ = Di >> 1, Ho = Hi >> 1, Wo = Wi >> 1;
    for (long e = (long)blockIdx.x * blockDim.x + threadIdx.x; e < total;
         e += (long)gridDim.x * blockDim.x) {
        int  c = (int)(e % C);
        long s = e / C;
        int  xo = (int)(s % Wo); s /= Wo;
        int  yo = (int)(s % Ho); s /= Ho;
        int  zo = (int)(s % Do_);
        int  b  = (int)(s / Do_);
        float m = -INFINITY;
        for (int dz = 0; dz < 3; ++dz) {
            int iz = 2 * zo + dz - 1; if (iz < 0 || iz >= Di) continue;
            for (int dy = 0; dy < 3; ++dy) {
                int iy = 2 * yo + dy - 1; if (iy < 0 || iy >= Hi) continue;
                for (int dx = 0; dx < 3; ++dx) {
                    int ix = 2 * xo + dx - 1; if (ix < 0 || ix >= Wi) continue;
                    float v = (float)in[((((long)b * Di + iz) * Hi + iy) * Wi + ix) * C + c];
                    m = fmaxf(m, v);
                }
            }
        }
        out[e] = (bf16_t)m;
    }
}

// ---------------------------------------------------------------------------
// Block 5 tail: normalize conv5 output, mean over the 8 spatial sites, ELU.
// ---------------------------------------------------------------------------
__launch_bounds__(256)
__global__ void finalize5_kernel(const float* __restrict__ f,
                                 const float* __restrict__ stats,
                                 float* __restrict__ h, int C, int Ns)
{
    int bc = blockIdx.x * 256 + threadIdx.x;
    if (bc >= 8 * C) return;
    int b = bc / C, c = bc - b * C;
    float mean = stats[2 * bc], rstd = stats[2 * bc + 1];
    float s = 0.f;
    for (int i = 0; i < Ns; ++i)
        s += (f[((long)b * Ns + i) * C + c] - mean) * rstd;
    s /= (float)Ns;
    h[bc] = s > 0.f ? s : expm1f(s);
}

// ---------------------------------------------------------------------------
// Final linear: out[b,n] = h[b,:] . w_lin[n,:] + b_lin[n]   (8x512x512, VALU)
// ---------------------------------------------------------------------------
__launch_bounds__(256)
__global__ void linear_kernel(const float* __restrict__ h,
                              const float* __restrict__ wl,
                              const float* __restrict__ bl,
                              float* __restrict__ out)
{
    int e = blockIdx.x * 256 + threadIdx.x;
    if (e >= 8 * 512) return;
    int b = e >> 9, n = e & 511;
    const float* hr = h + (long)b * 512;
    const float* wr = wl + (long)n * 512;
    float acc = bl[n];
    for (int k = 0; k < 512; ++k) acc = fmaf(hr[k], wr[k], acc);
    out[e] = acc;
}

// ---------------------------------------------------------------------------
extern "C" void kernel_launch(void* const* d_in, const int* in_sizes, int n_in,
                              void* d_out, int out_size, void* d_ws, size_t ws_size,
                              hipStream_t stream)
{
    (void)in_sizes; (void)n_in; (void)out_size; (void)ws_size;
    const float* x    = (const float*)d_in[0];
    const float* w1   = (const float*)d_in[1];
    const float* w2   = (const float*)d_in[2];
    const float* w3   = (const float*)d_in[3];
    const float* w4   = (const float*)d_in[4];
    const float* w5   = (const float*)d_in[5];
    const float* wlin = (const float*)d_in[6];
    const float* blin = (const float*)d_in[7];
    float* out = (float*)d_out;

    char* ws = (char*)d_ws;
    size_t off = 0;
    auto take = [&](size_t bytes) -> char* {
        off = (off + 255) & ~(size_t)255;
        char* pp = ws + off; off += bytes; return pp;
    };
    float*  F     = (float*) take(67108864);  // fp32 conv outputs (max: conv2)
    bf16_t* ACT0  = (bf16_t*)take(16777216);  // bf16 activations ping
    bf16_t* ACT1  = (bf16_t*)take(33554432);  // bf16 activations pong
    bf16_t* WP    = (bf16_t*)take(7077888);   // repacked weights (max: w5)
    float*  STATS = (float*) take(32768);     // per-(b,c) mean/rstd
    float*  H     = (float*) take(16384);     // pooled features [8,512]

    const int B = 8;

    // ---- block 1: conv s2 (direct) -> IN -> ELU -> bf16 ----
    {
        long total = (long)B * 32768 * 32;
        conv1_direct_kernel<<<(total + 255) / 256, 256, 0, stream>>>(x, w1, F, total);
        in_stats_kernel<<<B * 32, 256, 0, stream>>>(F, STATS, 32768, 32);
        norm_elu_bf16_kernel<<<(total + 255) / 256, 256, 0, stream>>>(F, STATS, ACT0,
                                                                      total, 32768, 32);
    }
    // ---- block 2: conv s1 (WMMA) -> IN -> ELU -> maxpool ----
    {
        long wtot = 27L * 1 * 4 * 256;
        repack_w_kernel<<<(wtot + 255) / 256, 256, 0, stream>>>(w2, WP, 32, 64, wtot);
        long tiles = ((long)B * 32768 / 16) * (64 / 32);
        conv3d_wmma_kernel<32, 32, 32, 32, 64, 1>
            <<<(tiles + 7) / 8, 256, 0, stream>>>(ACT0, WP, F, B);
        in_stats_kernel<<<B * 64, 256, 0, stream>>>(F, STATS, 32768, 64);
        long total = (long)B * 32768 * 64;
        norm_elu_bf16_kernel<<<(total + 255) / 256, 256, 0, stream>>>(F, STATS, ACT1,
                                                                      total, 32768, 64);
        long ptot = (long)B * 4096 * 64;
        maxpool3_bf16_kernel<<<(ptot + 255) / 256, 256, 0, stream>>>(
            ACT1, ACT0, B, 32, 32, 32, 64, ptot);
    }
    // ---- block 3 ----
    {
        long wtot = 27L * 2 * 8 * 256;
        repack_w_kernel<<<(wtot + 255) / 256, 256, 0, stream>>>(w3, WP, 64, 128, wtot);
        long tiles = ((long)B * 4096 / 16) * (128 / 32);
        conv3d_wmma_kernel<16, 16, 16, 64, 128, 1>
            <<<(tiles + 7) / 8, 256, 0, stream>>>(ACT0, WP, F, B);
        in_stats_kernel<<<B * 128, 256, 0, stream>>>(F, STATS, 4096, 128);
        long total = (long)B * 4096 * 128;
        norm_elu_bf16_kernel<<<(total + 255) / 256, 256, 0, stream>>>(F, STATS, ACT1,
                                                                      total, 4096, 128);
        long ptot = (long)B * 512 * 128;
        maxpool3_bf16_kernel<<<(ptot + 255) / 256, 256, 0, stream>>>(
            ACT1, ACT0, B, 16, 16, 16, 128, ptot);
    }
    // ---- block 4 ----
    {
        long wtot = 27L * 4 * 16 * 256;
        repack_w_kernel<<<(wtot + 255) / 256, 256, 0, stream>>>(w4, WP, 128, 256, wtot);
        long tiles = ((long)B * 512 / 16) * (256 / 32);
        conv3d_wmma_kernel<8, 8, 8, 128, 256, 1>
            <<<(tiles + 7) / 8, 256, 0, stream>>>(ACT0, WP, F, B);
        in_stats_kernel<<<B * 256, 256, 0, stream>>>(F, STATS, 512, 256);
        long total = (long)B * 512 * 256;
        norm_elu_bf16_kernel<<<(total + 255) / 256, 256, 0, stream>>>(F, STATS, ACT1,
                                                                      total, 512, 256);
        long ptot = (long)B * 64 * 256;
        maxpool3_bf16_kernel<<<(ptot + 255) / 256, 256, 0, stream>>>(
            ACT1, ACT0, B, 8, 8, 8, 256, ptot);
    }
    // ---- block 5: conv s2 (WMMA) -> IN -> global mean -> ELU ----
    {
        long wtot = 27L * 8 * 32 * 256;
        repack_w_kernel<<<(wtot + 255) / 256, 256, 0, stream>>>(w5, WP, 256, 512, wtot);
        long tiles = ((long)B * 8 / 16) * (512 / 32);
        conv3d_wmma_kernel<4, 4, 4, 256, 512, 2>
            <<<(tiles + 7) / 8, 256, 0, stream>>>(ACT0, WP, F, B);
        in_stats_kernel<<<B * 512, 256, 0, stream>>>(F, STATS, 8, 512);
        finalize5_kernel<<<16, 256, 0, stream>>>(F, STATS, H, 512, 8);
    }
    // ---- linear head ----
    linear_kernel<<<16, 256, 0, stream>>>(H, wlin, blin, out);
}
